// SelfAttention_1872605741484
// MI455X (gfx1250) — compile-verified, matrix operand order checked by reading, not run on hardware
//
#include <hip/hip_runtime.h>

typedef _Float16 half_t;
typedef __attribute__((ext_vector_type(16))) _Float16 v16h;
typedef __attribute__((ext_vector_type(8)))  _Float16 v8h;
typedef __attribute__((ext_vector_type(8)))  float    v8f;
typedef __attribute__((ext_vector_type(4)))  float    v4f;
typedef __attribute__((ext_vector_type(4)))  int      v4i;

constexpr int N_ = 2, L_ = 2048, E_ = 1024, H_ = 16, D_ = 64;
// softmax uses exp2: scale = log2(e)/sqrt(E) = 1.4426950408889634/32
constexpr float SCALE2  = 0.045084220027780106f;
constexpr float NEGMASK = -1e20f * 0.045084220027780106f;

#define WMMA(A, B, C) \
  __builtin_amdgcn_wmma_f32_16x16x32_f16(false, (A), false, (B), (short)0, (C), false, false)

__device__ __forceinline__ v8h load8f_cvt(const float* p) {
  v4f a = *(const v4f*)p;
  v4f b = *(const v4f*)(p + 4);
  v8h r;
  r[0] = (_Float16)a[0]; r[1] = (_Float16)a[1]; r[2] = (_Float16)a[2]; r[3] = (_Float16)a[3];
  r[4] = (_Float16)b[0]; r[5] = (_Float16)b[1]; r[6] = (_Float16)b[2]; r[7] = (_Float16)b[3];
  return r;
}
__device__ __forceinline__ v8h load8h(const half_t* p) { return *(const v8h*)p; }
__device__ __forceinline__ v16h cat(v8h a, v8h b) {
  return __builtin_shufflevector(a, b, 0,1,2,3,4,5,6,7,8,9,10,11,12,13,14,15);
}

// CDNA5 async copy: global -> LDS, 16B per lane, tracked by ASYNCcnt.
__device__ __forceinline__ void async_g2l_b128(uint32_t lds_byte_off, const void* gptr) {
  asm volatile("global_load_async_to_lds_b128 %0, %1, off"
               :: "v"(lds_byte_off), "v"((unsigned long long)(uintptr_t)gptr)
               : "memory");
}
__device__ __forceinline__ void wait_async0() {
  asm volatile("s_wait_asynccnt 0x0" ::: "memory");
}

// ---------------- Wo fp32 -> f16 ----------------
__global__ void cvt_f2h_kernel(const float* __restrict__ src, half_t* __restrict__ dst, int n) {
  int i = blockIdx.x * blockDim.x + threadIdx.x;
  if (i < n) dst[i] = (half_t)src[i];
}

// ---------------- Q/K projections: out[n,h,l,d] = x @ W^T ----------------
__global__ void proj_qk_kernel(const float* __restrict__ keysp,
                               const float* __restrict__ query,
                               const float* __restrict__ Wk,
                               const float* __restrict__ Wq,
                               half_t* __restrict__ qh,
                               half_t* __restrict__ kh) {
  const int lane = threadIdx.x & 31;
  const int wid  = blockIdx.x * (blockDim.x >> 5) + (threadIdx.x >> 5);
  const int lt    = wid & (L_ / 16 - 1);
  const int h     = (wid >> 7) & (H_ - 1);
  const int n     = (wid >> 11) & (N_ - 1);
  const int isk   = wid >> 12;                  // 0=q, 1=k (uniform scalar select)

  const float* x = isk ? keysp : query;
  const float* W = isk ? Wk    : Wq;
  half_t*    out = isk ? kh    : qh;

  const int lbase = lt * 16;
  const int col16 = lane & 15;
  const int kb    = (lane >> 4) * 8;

  const float* xrow = x + ((size_t)(n * L_ + lbase + col16) * E_ + h * D_);
  v16h xa0 = cat(load8f_cvt(xrow + kb),      load8f_cvt(xrow + kb + 16));
  v16h xa1 = cat(load8f_cvt(xrow + kb + 32), load8f_cvt(xrow + kb + 48));

  v8f c[4] = {};
#pragma unroll
  for (int t = 0; t < 4; ++t) {
    const float* wrow = W + (size_t)(t * 16 + col16) * D_;
    v16h wb0 = cat(load8f_cvt(wrow + kb),      load8f_cvt(wrow + kb + 16));
    v16h wb1 = cat(load8f_cvt(wrow + kb + 32), load8f_cvt(wrow + kb + 48));
    c[t] = WMMA(xa0, wb0, c[t]);
    c[t] = WMMA(xa1, wb1, c[t]);
  }

  const size_t nh = (size_t)(n * H_ + h);
#pragma unroll
  for (int t = 0; t < 4; ++t)
#pragma unroll
    for (int r = 0; r < 8; ++r)
      out[(nh * L_ + lbase + kb + r) * (size_t)D_ + t * 16 + col16] = (half_t)c[t][r];
}

// ---------------- V projection, stored transposed: vt[n,h,d,l] = Wv @ V^T ----------------
__global__ void proj_v_kernel(const float* __restrict__ values,
                              const float* __restrict__ Wv,
                              half_t* __restrict__ vt) {
  const int lane = threadIdx.x & 31;
  const int wid  = blockIdx.x * (blockDim.x >> 5) + (threadIdx.x >> 5);
  const int lt = wid & (L_ / 16 - 1);
  const int h  = (wid >> 7) & (H_ - 1);
  const int n  = wid >> 11;

  const int lbase = lt * 16;
  const int col16 = lane & 15;
  const int kb    = (lane >> 4) * 8;

  const float* xrow = values + ((size_t)(n * L_ + lbase + col16) * E_ + h * D_);
  v16h xb0 = cat(load8f_cvt(xrow + kb),      load8f_cvt(xrow + kb + 16));
  v16h xb1 = cat(load8f_cvt(xrow + kb + 32), load8f_cvt(xrow + kb + 48));

  v8f c[4] = {};
#pragma unroll
  for (int t = 0; t < 4; ++t) {
    const float* wrow = Wv + (size_t)(t * 16 + col16) * D_;
    v16h wa0 = cat(load8f_cvt(wrow + kb),      load8f_cvt(wrow + kb + 16));
    v16h wa1 = cat(load8f_cvt(wrow + kb + 32), load8f_cvt(wrow + kb + 48));
    c[t] = WMMA(wa0, xb0, c[t]);
    c[t] = WMMA(wa1, xb1, c[t]);
  }

  const size_t nh = (size_t)(n * H_ + h);
#pragma unroll
  for (int t = 0; t < 4; ++t)
#pragma unroll
    for (int r = 0; r < 8; ++r)
      vt[(nh * D_ + t * 16 + kb + r) * (size_t)L_ + lbase + col16] = (half_t)c[t][r];
}

// ---------------- flash attention ----------------
// Block = 8 waves, one (n,h), 8 adjacent 16-query tiles. K/V tiles for each
// 32-key step are staged into LDS with async global->LDS copies (double
// buffered); each wave issues 1/8 of the next tile, computes from the current.
__global__ void __launch_bounds__(256) attn_kernel(const half_t* __restrict__ qh,
                                                   const half_t* __restrict__ kh,
                                                   const half_t* __restrict__ vt,
                                                   const int*    __restrict__ mask,
                                                   half_t*       __restrict__ att) {
  __shared__ __align__(16) half_t skbuf[2][32 * 64];  // K rows [k-local][d]
  __shared__ __align__(16) half_t svbuf[2][64 * 32];  // V^T rows [d][k-local]

  const int lane = threadIdx.x & 31;
  const int wave = threadIdx.x >> 5;                  // 0..7
  const int qb = blockIdx.x & (L_ / 128 - 1);         // 16 query blocks
  const int h  = (blockIdx.x >> 4) & (H_ - 1);
  const int n  = blockIdx.x >> 8;
  const int qt = qb * 8 + wave;

  const int col16 = lane & 15;
  const int kb    = (lane >> 4) * 8;
  const int qrow  = qt * 16 + col16;
  const size_t nh = (size_t)(n * H_ + h);

  // staging source pointers (bytes)
  const char* kg = (const char*)(kh + nh * (size_t)L_ * D_);
  const char* vg = (const char*)(vt + nh * (size_t)D_ * L_);
  // this wave's slice: K chunk = 512B at wave*512; V chunk = rows d=wave*8..+7
  const int vrow   = wave * 8 + (lane >> 2);          // d row for V staging
  const int vcol16 = (lane & 3) * 16;                 // byte offset within 64B row

  // Q as B operand of S^T = K * Q^T
  const half_t* qp = qh + (nh * L_ + qrow) * D_;
  v16h bq0 = cat(load8h(qp + kb),      load8h(qp + kb + 16));
  v16h bq1 = cat(load8h(qp + kb + 32), load8h(qp + kb + 48));

  float m = -3.0e38f, l = 0.0f;
  v8f o[4] = {};   // O^T accumulators: tile t holds rows d = t*16+kb+r, col q = col16

  const int* mrow = mask + ((size_t)n * L_ + qrow) * L_;

  // prologue: stage buffer 0 (kk = 0)
  {
    async_g2l_b128((uint32_t)(uintptr_t)&skbuf[0][wave * 256 + lane * 8],
                   kg + wave * 512 + lane * 16);
    async_g2l_b128((uint32_t)(uintptr_t)&svbuf[0][vrow * 32 + (lane & 3) * 8],
                   vg + (size_t)vrow * L_ * 2 + vcol16);
    wait_async0();
    __syncthreads();
  }

  for (int i = 0; i < L_ / 32; ++i) {
    const int buf = i & 1;
    const int kk  = i * 32;

    // issue async staging of the next tile into the other buffer
    if (i + 1 < L_ / 32) {
      const int kn = kk + 32;
      async_g2l_b128((uint32_t)(uintptr_t)&skbuf[buf ^ 1][wave * 256 + lane * 8],
                     kg + (size_t)kn * 128 + wave * 512 + lane * 16);
      async_g2l_b128((uint32_t)(uintptr_t)&svbuf[buf ^ 1][vrow * 32 + (lane & 3) * 8],
                     vg + ((size_t)vrow * L_ + kn) * 2 + vcol16);
    }

    // ---- S^T tiles from LDS (keys kk..kk+15 and kk+16..kk+31) ----
    const half_t* kr0 = &skbuf[buf][col16 * 64];
    const half_t* kr1 = &skbuf[buf][(16 + col16) * 64];
    v8f st0 = {}, st1 = {};
    v16h ak;
    ak  = cat(load8h(kr0 + kb),      load8h(kr0 + kb + 16));
    st0 = WMMA(ak, bq0, st0);
    ak  = cat(load8h(kr0 + kb + 32), load8h(kr0 + kb + 48));
    st0 = WMMA(ak, bq1, st0);
    ak  = cat(load8h(kr1 + kb),      load8h(kr1 + kb + 16));
    st1 = WMMA(ak, bq0, st1);
    ak  = cat(load8h(kr1 + kb + 32), load8h(kr1 + kb + 48));
    st1 = WMMA(ak, bq1, st1);

    // ---- mask + scale into base-2 domain ----
    v4i m0a = *(const v4i*)(mrow + kk + kb);
    v4i m0b = *(const v4i*)(mrow + kk + kb + 4);
    v4i m1a = *(const v4i*)(mrow + kk + 16 + kb);
    v4i m1b = *(const v4i*)(mrow + kk + 16 + kb + 4);
    float s0[8], s1[8];
#pragma unroll
    for (int r = 0; r < 4; ++r) {
      s0[r]     = m0a[r] ? st0[r]     * SCALE2 : NEGMASK;
      s0[r + 4] = m0b[r] ? st0[r + 4] * SCALE2 : NEGMASK;
      s1[r]     = m1a[r] ? st1[r]     * SCALE2 : NEGMASK;
      s1[r + 4] = m1b[r] ? st1[r + 4] * SCALE2 : NEGMASK;
    }

    // ---- online softmax (per-lane q column; partner lane is lane^16) ----
    float tmax = s0[0];
#pragma unroll
    for (int r = 1; r < 8; ++r) tmax = fmaxf(tmax, s0[r]);
#pragma unroll
    for (int r = 0; r < 8; ++r) tmax = fmaxf(tmax, s1[r]);
    tmax = fmaxf(tmax, __shfl_xor(tmax, 16));
    const float mnew  = fmaxf(m, tmax);
    const float alpha = exp2f(m - mnew);

    float psum = 0.0f;
    v16h bp;                 // P^T as B operand: halves 0..7 <- st0, 8..15 <- st1
#pragma unroll
    for (int r = 0; r < 8; ++r) {
      float p0 = exp2f(s0[r] - mnew); psum += p0; bp[r]     = (_Float16)p0;
      float p1 = exp2f(s1[r] - mnew); psum += p1; bp[r + 8] = (_Float16)p1;
    }
    psum += __shfl_xor(psum, 16);
    l = l * alpha + psum;
    m = mnew;

    // ---- rescale + O^T += V^T * P^T (V^T from LDS) ----
#pragma unroll
    for (int t = 0; t < 4; ++t) {
#pragma unroll
      for (int r = 0; r < 8; ++r) o[t][r] *= alpha;
      const half_t* vr = &svbuf[buf][(t * 16 + col16) * 32];
      v16h av = cat(load8h(vr + kb), load8h(vr + 16 + kb));
      o[t] = WMMA(av, bp, o[t]);
    }

    // publish next buffer / retire reads of current one
    wait_async0();
    __syncthreads();
  }

  // ---- normalize and store att[n, q, h*64 + d] (f16) ----
  const float invl = 1.0f / l;
  half_t* op = att + ((size_t)(n * L_ + qrow)) * E_ + h * D_ + kb;
#pragma unroll
  for (int t = 0; t < 4; ++t) {
    v8h hv;
#pragma unroll
    for (int r = 0; r < 8; ++r) hv[r] = (_Float16)(o[t][r] * invl);
    *(v8h*)(op + t * 16) = hv;
  }
}

// ---------------- output projection: Y = att @ Wo^T + bo (fp32 out) ----------------
__global__ void outproj_kernel(const half_t* __restrict__ att,
                               const half_t* __restrict__ WoH,
                               const float*  __restrict__ bo,
                               float*        __restrict__ out) {
  const int lane = threadIdx.x & 31;
  const int wid  = blockIdx.x * (blockDim.x >> 5) + (threadIdx.x >> 5);
  const int ct = wid & 15;        // 16 column groups of 64
  const int mt = wid >> 4;        // 256 row tiles of 16
  const int col16 = lane & 15;
  const int kb    = (lane >> 4) * 8;

  const half_t* arow = att + (size_t)(mt * 16 + col16) * E_;
  v8f c[4] = {};
  for (int ks = 0; ks < E_; ks += 32) {
    v16h a = cat(load8h(arow + ks + kb), load8h(arow + ks + kb + 16));
#pragma unroll
    for (int t = 0; t < 4; ++t) {
      const half_t* wrow = WoH + (size_t)(ct * 64 + t * 16 + col16) * E_ + ks;
      v16h b = cat(load8h(wrow + kb), load8h(wrow + kb + 16));
      c[t] = WMMA(a, b, c[t]);
    }
  }
#pragma unroll
  for (int t = 0; t < 4; ++t) {
    const int col   = ct * 64 + t * 16 + col16;
    const float bias = bo[col];
#pragma unroll
    for (int r = 0; r < 8; ++r)
      out[(size_t)(mt * 16 + kb + r) * E_ + col] = c[t][r] + bias;
  }
}

extern "C" void kernel_launch(void* const* d_in, const int* in_sizes, int n_in,
                              void* d_out, int out_size, void* d_ws, size_t ws_size,
                              hipStream_t stream) {
  (void)in_sizes; (void)n_in; (void)out_size; (void)ws_size;
  const float* values = (const float*)d_in[0];
  const float* keysp  = (const float*)d_in[1];
  const float* query  = (const float*)d_in[2];
  const int*   mask   = (const int*)  d_in[3];
  const float* Wv     = (const float*)d_in[4];
  const float* Wk     = (const float*)d_in[5];
  const float* Wq     = (const float*)d_in[6];
  const float* Wo     = (const float*)d_in[7];
  const float* bo     = (const float*)d_in[8];

  constexpr size_t QKV = (size_t)N_ * H_ * L_ * D_;   // 4,194,304 halfs each
  constexpr size_t NLE = (size_t)N_ * L_ * E_;        // 4,194,304 halfs
  half_t* qh  = (half_t*)d_ws;
  half_t* kh  = qh  + QKV;
  half_t* vt  = kh  + QKV;
  half_t* att = vt  + QKV;
  half_t* WoH = att + NLE;                            // E*E halfs; ~34 MB total

  cvt_f2h_kernel<<<(E_ * E_) / 256, 256, 0, stream>>>(Wo, WoH, E_ * E_);
  proj_qk_kernel<<<2 * N_ * H_ * (L_ / 16) / 4, 128, 0, stream>>>(
      keysp, query, Wk, Wq, qh, kh);
  proj_v_kernel<<<N_ * H_ * (L_ / 16) / 4, 128, 0, stream>>>(values, Wv, vt);
  attn_kernel<<<N_ * H_ * (L_ / 128), 256, 0, stream>>>(qh, kh, vt, mask, att);
  outproj_kernel<<<(N_ * L_ / 16) * (E_ / 64) / 4, 128, 0, stream>>>(
      att, WoH, bo, (float*)d_out);
}
